// GraphEmbedder_87763361726596
// MI455X (gfx1250) — matrix operand order repaired and checked by reading, not compiled, not run on hardware
//
#include <hip/hip_runtime.h>

#define N_NODES 50000
#define N_EDGES 800000
#define N_GRAPHS 64
#define D_IN 128
#define D_HID 64
#define D_OUT 32

typedef __attribute__((ext_vector_type(16))) _Float16 v16h;
typedef __attribute__((ext_vector_type(8)))  _Float16 v8h;
typedef __attribute__((ext_vector_type(4)))  _Float16 v4h;
typedef __attribute__((ext_vector_type(8)))  float    v8f;

// ---------------------------------------------------------------------------
// prep: W -> f16 transposed (Bt[n][k]), zero pool/counts, deg = 1 (self loop)
// ---------------------------------------------------------------------------
__global__ void k_prep(const float* __restrict__ W1, const float* __restrict__ W2,
                       _Float16* __restrict__ W1t, _Float16* __restrict__ W2t,
                       float* __restrict__ pool, float* __restrict__ counts,
                       float* __restrict__ deg) {
  int tid = blockIdx.x * blockDim.x + threadIdx.x;
  int stride = gridDim.x * blockDim.x;
  for (int i = tid; i < D_HID * D_IN; i += stride) {
    int n = i / D_IN, k = i % D_IN;
    W1t[i] = (_Float16)W1[k * D_HID + n];
  }
  for (int i = tid; i < D_OUT * D_HID; i += stride) {
    int n = i / D_HID, k = i % D_HID;
    W2t[i] = (_Float16)W2[k * D_OUT + n];
  }
  for (int i = tid; i < N_GRAPHS * D_OUT; i += stride) pool[i] = 0.0f;
  for (int i = tid; i < N_GRAPHS; i += stride) counts[i] = 0.0f;
  for (int i = tid; i < N_NODES; i += stride) deg[i] = 1.0f;
}

__global__ void k_deg(const int* __restrict__ dst, float* __restrict__ deg) {
  int e = blockIdx.x * blockDim.x + threadIdx.x;
  if (e < N_EDGES) atomicAdd(&deg[dst[e]], 1.0f);
}

__global__ void k_dinv(float* __restrict__ deg) {
  int n = blockIdx.x * blockDim.x + threadIdx.x;
  if (n < N_NODES) deg[n] = rsqrtf(deg[n]);   // deg >= 1 always (self loop)
}

// ---------------------------------------------------------------------------
// row-normalize x (sum clamped >= 1), emit f16 features. One wave per row.
// ---------------------------------------------------------------------------
__global__ void k_rownorm(const float* __restrict__ x, _Float16* __restrict__ xh) {
  int gid = blockIdx.x * blockDim.x + threadIdx.x;
  int row = gid >> 5;
  int lane = threadIdx.x & 31;
  if (row >= N_NODES) return;
  const float4 f = ((const float4*)(x + (size_t)row * D_IN))[lane];
  float s = f.x + f.y + f.z + f.w;
  #pragma unroll
  for (int o = 16; o > 0; o >>= 1) s += __shfl_xor(s, o, 32);
  float sc = 1.0f / fmaxf(s, 1.0f);
  v4h h;
  h[0] = (_Float16)(f.x * sc);
  h[1] = (_Float16)(f.y * sc);
  h[2] = (_Float16)(f.z * sc);
  h[3] = (_Float16)(f.w * sc);
  ((v4h*)(xh + (size_t)row * D_IN))[lane] = h;
}

// ---------------------------------------------------------------------------
// WMMA GEMM: D[16 x NT*16] = A[16 x KDIM](f16) * Bt(f16, [n][k] layout), f32 acc
// One wave per 16-row tile. Layouts used here (wave32; A/B share a contiguous k order):
//   A 16-bit:  lane half g: elems e -> K = k0 + 16g + e
//   B 16-bit:  N = lane&15, elems e -> K = k0 + 16g + e  (contiguous in Bt row)
//   C/D f32:   VGPR r: M = 8g + r, N = lane&15
// ---------------------------------------------------------------------------
template<int KDIM, int NT>
__global__ __launch_bounds__(32)
void k_gemm(const _Float16* __restrict__ A, const _Float16* __restrict__ Bt,
            float* __restrict__ D) {
  const int lane = threadIdx.x & 31;
  const int g = lane >> 4;
  const int l = lane & 15;
  const int row = blockIdx.x * 16 + l;
  v8f acc[NT] = {};
  const _Float16* arow = A + (size_t)row * KDIM;
  #pragma unroll
  for (int k0 = 0; k0 < KDIM; k0 += 32) {
    v16h a = *(const v16h*)(arow + k0 + g * 16);
    #pragma unroll
    for (int t = 0; t < NT; ++t) {
      v16h b = *(const v16h*)(Bt + (size_t)(t * 16 + l) * KDIM + k0 + g * 16);
      acc[t] = __builtin_amdgcn_wmma_f32_16x16x32_f16(
          false, a, false, b, (short)0, acc[t], false, false);
    }
  }
  const int N = NT * 16;
  #pragma unroll
  for (int t = 0; t < NT; ++t) {
    #pragma unroll
    for (int r = 0; r < 8; ++r) {
      D[(size_t)(blockIdx.x * 16 + g * 8 + r) * N + t * 16 + l] = acc[t][r];
    }
  }
}

// out[n][c] = b[c] + h[n][c] * dinv[n]^2   (bias + self-loop contribution)
__global__ void k_bias_self(const float* __restrict__ h, const float* __restrict__ dinv,
                            const float* __restrict__ b, float* __restrict__ o,
                            int logC, int total) {
  int idx = blockIdx.x * blockDim.x + threadIdx.x;
  if (idx >= total) return;
  int n = idx >> logC;
  int c = idx & ((1 << logC) - 1);
  float di = dinv[n];
  o[idx] = b[c] + h[idx] * (di * di);
}

// edge scatter: (1<<logQ) threads per edge, each handles a float4 of the row.
// h row stride = C floats. out gets atomic f32 adds (rows stay L2 resident).
__global__ void k_scatter(const int* __restrict__ src, const int* __restrict__ dst,
                          const float* __restrict__ dinv, const float* __restrict__ h,
                          float* __restrict__ o, int logQ, int C) {
  int tid = blockIdx.x * blockDim.x + threadIdx.x;
  int e = tid >> logQ;
  if (e >= N_EDGES) return;
  int q = tid & ((1 << logQ) - 1);
  int s = src[e], d = dst[e];
  float nw = dinv[s] * dinv[d];
  const float4 v = ((const float4*)(h + (size_t)s * C))[q];
  float* op = o + (size_t)d * C + q * 4;
  atomicAdd(op + 0, v.x * nw);
  atomicAdd(op + 1, v.y * nw);
  atomicAdd(op + 2, v.z * nw);
  atomicAdd(op + 3, v.w * nw);
}

__global__ void k_relu_f16(const float* __restrict__ in, _Float16* __restrict__ out,
                           int total) {
  int idx = blockIdx.x * blockDim.x + threadIdx.x;
  if (idx < total) out[idx] = (_Float16)fmaxf(in[idx], 0.0f);
}

// relu + atomic mean-pool accumulation
__global__ void k_pool(const float* __restrict__ h, const int* __restrict__ batch,
                       float* __restrict__ pool, float* __restrict__ counts) {
  int idx = blockIdx.x * blockDim.x + threadIdx.x;
  if (idx >= N_NODES * D_OUT) return;
  int n = idx >> 5, c = idx & 31;
  int gph = batch[n];
  atomicAdd(&pool[gph * D_OUT + c], fmaxf(h[idx], 0.0f));
  if (c == 0) atomicAdd(&counts[gph], 1.0f);
}

__global__ void k_final(const float* __restrict__ pool, const float* __restrict__ counts,
                        float* __restrict__ out) {
  int idx = blockIdx.x * blockDim.x + threadIdx.x;
  if (idx >= N_GRAPHS * D_OUT) return;
  out[idx] = pool[idx] / fmaxf(counts[idx >> 5], 1.0f);
}

// ---------------------------------------------------------------------------
extern "C" void kernel_launch(void* const* d_in, const int* in_sizes, int n_in,
                              void* d_out, int out_size, void* d_ws, size_t ws_size,
                              hipStream_t stream) {
  (void)in_sizes; (void)n_in; (void)out_size; (void)ws_size;
  const float* x   = (const float*)d_in[0];
  const int*   ei  = (const int*)d_in[1];
  const int*   bat = (const int*)d_in[2];
  const float* W1  = (const float*)d_in[3];
  const float* b1  = (const float*)d_in[4];
  const float* W2  = (const float*)d_in[5];
  const float* b2  = (const float*)d_in[6];
  float* out = (float*)d_out;
  const int* src = ei;
  const int* dst = ei + N_EDGES;

  char* w = (char*)d_ws;
  size_t off = 0;
  auto take = [&](size_t bytes) -> char* {
    char* p = w + off;
    off += (bytes + 255) & ~(size_t)255;
    return p;
  };
  float*    dinv = (float*)take((size_t)N_NODES * 4);
  _Float16* xh   = (_Float16*)take((size_t)N_NODES * D_IN * 2);  // reused: h1r (f16)
  float*    bufA = (float*)take((size_t)N_NODES * D_HID * 4);    // h1, later out2
  float*    bufB = (float*)take((size_t)N_NODES * D_HID * 4);    // out1, later h2
  _Float16* W1t  = (_Float16*)take((size_t)D_HID * D_IN * 2);
  _Float16* W2t  = (_Float16*)take((size_t)D_OUT * D_HID * 2);
  float*    pool = (float*)take((size_t)N_GRAPHS * D_OUT * 4);
  float*    cnts = (float*)take((size_t)N_GRAPHS * 4);

  float*    h1   = bufA;
  float*    out1 = bufB;
  _Float16* h1r  = xh;     // xh dead after gemm1
  float*    h2   = bufB;   // out1 dead after relu->f16
  float*    out2 = bufA;   // h1 dead after scatter1

  k_prep<<<196, 256, 0, stream>>>(W1, W2, W1t, W2t, pool, cnts, dinv);
  k_deg<<<(N_EDGES + 255) / 256, 256, 0, stream>>>(dst, dinv);
  k_dinv<<<(N_NODES + 255) / 256, 256, 0, stream>>>(dinv);
  k_rownorm<<<(N_NODES * 32 + 255) / 256, 256, 0, stream>>>(x, xh);

  // layer 1: h1 = xn @ W1  (50000 = 3125 * 16 exact, EXEC all-ones per wave)
  k_gemm<D_IN, 4><<<N_NODES / 16, 32, 0, stream>>>(xh, W1t, h1);
  k_bias_self<<<(N_NODES * D_HID + 255) / 256, 256, 0, stream>>>(
      h1, dinv, b1, out1, 6, N_NODES * D_HID);
  k_scatter<<<(N_EDGES * 16 + 255) / 256, 256, 0, stream>>>(
      src, dst, dinv, h1, out1, 4, D_HID);
  k_relu_f16<<<(N_NODES * D_HID + 255) / 256, 256, 0, stream>>>(
      out1, h1r, N_NODES * D_HID);

  // layer 2: h2 = relu(out1) @ W2
  k_gemm<D_HID, 2><<<N_NODES / 16, 32, 0, stream>>>(h1r, W2t, h2);
  k_bias_self<<<(N_NODES * D_OUT + 255) / 256, 256, 0, stream>>>(
      h2, dinv, b2, out2, 5, N_NODES * D_OUT);
  k_scatter<<<(N_EDGES * 8 + 255) / 256, 256, 0, stream>>>(
      src, dst, dinv, h2, out2, 3, D_OUT);

  // global mean pool
  k_pool<<<(N_NODES * D_OUT + 255) / 256, 256, 0, stream>>>(out2, bat, pool, cnts);
  k_final<<<(N_GRAPHS * D_OUT + 255) / 256, 256, 0, stream>>>(pool, cnts, out);
}
